// TextLSTMModule_30485677867901
// MI455X (gfx1250) — matrix operand order repaired and checked by reading, not compiled, not run on hardware
//
#include <hip/hip_runtime.h>
#include <stdint.h>

// ---------------------------------------------------------------------------
// LSTM encoder for MI455X (gfx1250, wave32, WMMA).
// Phase 1: embed + fp32->bf16 weight conversion (bandwidth trivial).
// Phase 2: Gx = X @ W_ih^T + (b_ih+b_hh) as a bf16 WMMA GEMM (17.2 GFLOP).
// Phase 3: persistent 32-WG kernel; per step:
//   - h (2 KB) staged into LDS via global_load_async_to_lds_b128 (ASYNCcnt),
//   - h@W_hh^T via WMMA (A = W_hh row tiles from L2, B = h broadcast from LDS),
//   - fused gate nonlinearities (c/h live in registers),
//   - monotonic global-atomic grid barrier (h double-buffered in bf16).
// Phase 3 is L2-bandwidth bound (8 MB of W_hh per step); 32 WGs halve the
// per-WGP traffic vs 16 while keeping barrier cost negligible.
// ---------------------------------------------------------------------------

#define S_LEN 2048
#define E_DIM 1024
#define H_DIM 1024
#define R_DIM 4096   // 4*H
#define NWG   32     // persistent workgroups in the recurrent phase

typedef __attribute__((ext_vector_type(16))) __bf16 v16bf;
typedef __attribute__((ext_vector_type(8)))  float  v8f;

struct __attribute__((aligned(16))) U128 { uint32_t a, b, c, d; };
union Frag16 { v16bf bf; U128 q[2]; };

// A-matrix (16x32 bf16, MxK): lane = row M; lanes 0-15 hold K0..15,
// lanes 16-31 hold K16..31 (contiguous per lane-half, matching B).
__device__ __forceinline__ v16bf frag_a_rows(const uint16_t* base, int ld) {
  const int lane = threadIdx.x & 31;
  const int l16  = lane & 15;
  const int hi   = lane >> 4;
  const uint16_t* p = base + (size_t)l16 * ld + hi * 16;
  Frag16 f;
  f.q[0] = *(const U128*)(p);
  f.q[1] = *(const U128*)(p + 8);
  return f.bf;
}

// B-matrix (32x16 bf16, KxN): lane = column N; lanes 0-15 hold K0..15,
// lanes 16-31 hold K16..31 (contiguous per lane-half).
__device__ __forceinline__ v16bf frag_b_cols(const uint16_t* base, int ld) {
  const int lane = threadIdx.x & 31;
  const int l16  = lane & 15;
  const int hi   = lane >> 4;
  const uint16_t* p = base + (size_t)l16 * ld + hi * 16;
  Frag16 f;
  f.q[0] = *(const U128*)(p);
  f.q[1] = *(const U128*)(p + 8);
  return f.bf;
}

// B = vector v broadcast into all 16 columns (matvec through the matrix pipe).
// v points into LDS -> compiler emits ds_load_b128.
__device__ __forceinline__ v16bf frag_b_vec(const uint16_t* v) {
  const int hi = (threadIdx.x >> 4) & 1;
  const uint16_t* p = v + hi * 16;
  Frag16 f;
  f.q[0] = *(const U128*)(p);
  f.q[1] = *(const U128*)(p + 8);
  return f.bf;
}

__device__ __forceinline__ v8f wmma_bf16(v16bf a, v16bf b, v8f c) {
  // (neg_a, A, neg_b, B, c_mod, C, reuse_a, reuse_b)
  return __builtin_amdgcn_wmma_f32_16x16x32_bf16(false, a, false, b,
                                                 (short)0, c, false, false);
}

__device__ __forceinline__ uint16_t f2bf(float x) {  // RNE fp32->bf16
  union { float f; uint32_t u; } v; v.f = x;
  uint32_t r = v.u + 0x7FFFu + ((v.u >> 16) & 1u);
  return (uint16_t)(r >> 16);
}

__device__ __forceinline__ float sigf(float x) { return 1.0f / (1.0f + __expf(-x)); }

// ------------------------------ Phase 1 ------------------------------------

__global__ void lstm_prep_weights(const float* __restrict__ Wih,
                                  const float* __restrict__ Whh,
                                  const float* __restrict__ b_ih,
                                  const float* __restrict__ b_hh,
                                  uint16_t* __restrict__ Wihb,
                                  uint16_t* __restrict__ Whhb,
                                  float* __restrict__ bias) {
  const int i = blockIdx.x * 256 + threadIdx.x;   // exactly R_DIM*E_DIM threads
  Wihb[i] = f2bf(Wih[i]);
  Whhb[i] = f2bf(Whh[i]);
  if (i < R_DIM) bias[i] = b_ih[i] + b_hh[i];
}

__global__ void lstm_embed(const int* __restrict__ tokens,
                           const float* __restrict__ emb,
                           uint16_t* __restrict__ Xbf) {
  const int s   = blockIdx.x;
  const int tok = tokens[s];
  const float*   src = emb + (size_t)tok * E_DIM;
  uint16_t*      dst = Xbf + (size_t)s * E_DIM;
  #pragma unroll
  for (int j = 0; j < 4; ++j) {
    const int e = threadIdx.x + j * 256;
    dst[e] = f2bf(src[e]);
  }
}

__global__ void lstm_init(uint16_t* __restrict__ hbuf, unsigned* __restrict__ cnt) {
  #pragma unroll
  for (int j = 0; j < 8; ++j) hbuf[threadIdx.x + j * 256] = 0;  // both h buffers
  if (threadIdx.x == 0) *cnt = 0u;
}

// ------------------------------ Phase 2 ------------------------------------
// Gx[s,r] = sum_e X[s,e]*Wih[r,e] + bias[r].  Per-wave 32x64 tile (2x4 WMMA),
// per-block 64x256, grid (4096/256, 2048/64) = (16, 32).

__global__ __launch_bounds__(256) void lstm_gemm_x(
    const uint16_t* __restrict__ Xbf, const uint16_t* __restrict__ Wihb,
    const float* __restrict__ bias, float* __restrict__ Gx) {
  const int wave  = threadIdx.x >> 5;
  const int lane  = threadIdx.x & 31;
  const int l16   = lane & 15;
  const int hi    = lane >> 4;
  const int m0    = blockIdx.y * 64 + (wave >> 2) * 32;   // sequence rows
  const int n0    = blockIdx.x * 256 + (wave & 3) * 64;   // gate rows (output cols)

  const v8f z = {0.f, 0.f, 0.f, 0.f, 0.f, 0.f, 0.f, 0.f};
  v8f acc[2][4];
  #pragma unroll
  for (int i = 0; i < 2; ++i)
    #pragma unroll
    for (int j = 0; j < 4; ++j) acc[i][j] = z;

  for (int k0 = 0; k0 < E_DIM; k0 += 32) {
    v16bf a[2], b[4];
    #pragma unroll
    for (int i = 0; i < 2; ++i)
      a[i] = frag_a_rows(Xbf + (size_t)(m0 + 16 * i) * E_DIM + k0, E_DIM);
    #pragma unroll
    for (int j = 0; j < 4; ++j)
      b[j] = frag_b_cols(Wihb + (size_t)(n0 + 16 * j) * E_DIM + k0, E_DIM);
    #pragma unroll
    for (int i = 0; i < 2; ++i)
      #pragma unroll
      for (int j = 0; j < 4; ++j) acc[i][j] = wmma_bf16(a[i], b[j], acc[i][j]);
  }

  #pragma unroll
  for (int i = 0; i < 2; ++i) {
    #pragma unroll
    for (int j = 0; j < 4; ++j) {
      const int n  = n0 + 16 * j + l16;
      const float bv = bias[n];
      #pragma unroll
      for (int r = 0; r < 8; ++r) {
        const int m = m0 + 16 * i + hi * 8 + r;   // C/D layout: row = r + 8*hi
        Gx[(size_t)m * R_DIM + n] = acc[i][j][r] + bv;
      }
    }
  }
}

// ------------------------------ Phase 3 ------------------------------------
// Persistent: 32 WGs x 256 threads. WG `blk` owns hidden units
// [blk*32, blk*32+32) across all 4 gates (128 rows of W_hh = 8 row-tiles,
// one per wave). h is staged into LDS each step with async-to-LDS loads.

__global__ __launch_bounds__(256) void lstm_recurrent(
    const uint16_t* __restrict__ Whhb, const float* __restrict__ Gx,
    uint16_t* __restrict__ hbuf, unsigned* __restrict__ cnt,
    float* __restrict__ out) {
  __shared__ __align__(16) uint16_t lds_h[H_DIM];   // 2 KB: current h in bf16
  __shared__ float lds_gate[128];                   // W_hh dot results
  const int tid  = threadIdx.x;
  const int wave = tid >> 5;
  const int lane = tid & 31;
  const int l16  = lane & 15;
  const int hi   = lane >> 4;
  const int blk  = blockIdx.x;
  const v8f z = {0.f, 0.f, 0.f, 0.f, 0.f, 0.f, 0.f, 0.f};

  float cval = 0.0f, hval = 0.0f;   // cell/hidden state lives in registers

  for (int t = 0; t < S_LEN; ++t) {
    const uint16_t* hread = hbuf + (size_t)(t & 1) * H_DIM;

    // ---- Stage h (1024 bf16 = 2 KB) into LDS: 128 async b128 copies. ----
    if (tid < 128) {
      const uint32_t dst = (uint32_t)(uintptr_t)&lds_h[tid * 8];
      const uint64_t src = (uint64_t)(uintptr_t)(hread + tid * 8);
      asm volatile("global_load_async_to_lds_b128 %0, %1, off"
                   :: "v"(dst), "v"(src) : "memory");
    }
    asm volatile("s_wait_asynccnt 0x0" ::: "memory");
    __syncthreads();

    // ---- Wave `wave` owns row-tile tt = wave (16 rows of W_hh). ----
    {
      const int tt = wave;
      const int g  = tt >> 1;                              // gate 0..3
      const int rb = g * H_DIM + blk * 32 + (tt & 1) * 16; // global row base
      v8f acc = z;
      for (int k0 = 0; k0 < H_DIM; k0 += 32) {
        const v16bf a = frag_a_rows(Whhb + (size_t)rb * H_DIM + k0, H_DIM);
        const v16bf b = frag_b_vec(&lds_h[k0]);            // ds_load_b128
        acc = wmma_bf16(a, b, acc);
      }
      // Column 0 of D holds dot(W row, h); lanes 0 and 16 own rows 0-7 / 8-15.
      if (l16 == 0) {
        const int lr = tt * 16 + hi * 8;
        #pragma unroll
        for (int r = 0; r < 8; ++r) lds_gate[lr + r] = acc[r];
      }
    }
    __syncthreads();

    if (tid < 32) {
      const int gu = blk * 32 + tid;
      const float* gx = Gx + (size_t)t * R_DIM;   // includes b_ih + b_hh
      const float gi = gx[gu]             + lds_gate[tid];
      const float gf = gx[H_DIM + gu]     + lds_gate[32 + tid];
      const float gg = gx[2 * H_DIM + gu] + lds_gate[64 + tid];
      const float go = gx[3 * H_DIM + gu] + lds_gate[96 + tid];
      cval = sigf(gf) * cval + sigf(gi) * tanhf(gg);
      hval = sigf(go) * tanhf(cval);
      hbuf[(size_t)((t + 1) & 1) * H_DIM + gu] = f2bf(hval);
      if (t + 1 < S_LEN) __builtin_prefetch(gx + R_DIM + gu, 0, 1);
    }
    __syncthreads();      // lds reads done before next step's writes
    __threadfence();      // make h writes device-visible before barrier arrive

    if (tid == 0) {
      atomicAdd(cnt, 1u);                       // monotonic: no reset races
      const unsigned target = (unsigned)NWG * (unsigned)(t + 1);
      while (__hip_atomic_load(cnt, __ATOMIC_ACQUIRE, __HIP_MEMORY_SCOPE_AGENT)
             < target) {
        __builtin_amdgcn_s_sleep(2);
      }
    }
    __syncthreads();
  }

  if (tid < 32) out[blk * 32 + tid] = hval;   // final hidden state [1,1024]
}

// ------------------------------ Launch -------------------------------------

extern "C" void kernel_launch(void* const* d_in, const int* in_sizes, int n_in,
                              void* d_out, int out_size, void* d_ws, size_t ws_size,
                              hipStream_t stream) {
  (void)in_sizes; (void)n_in; (void)out_size; (void)ws_size;
  const int*   tokens = (const int*)d_in[0];
  const float* emb    = (const float*)d_in[1];
  const float* Wih    = (const float*)d_in[2];
  const float* Whh    = (const float*)d_in[3];
  const float* b_ih   = (const float*)d_in[4];
  const float* b_hh   = (const float*)d_in[5];
  float* out = (float*)d_out;

  char* w = (char*)d_ws;
  size_t off = 0;
  auto carve = [&](size_t bytes) -> void* {
    void* p = w + off;
    off = (off + bytes + 255) & ~(size_t)255;
    return p;
  };
  uint16_t* Xbf   = (uint16_t*)carve((size_t)S_LEN * E_DIM * 2);   //  4 MiB
  uint16_t* Wihb  = (uint16_t*)carve((size_t)R_DIM * E_DIM * 2);   //  8 MiB
  uint16_t* Whhb  = (uint16_t*)carve((size_t)R_DIM * H_DIM * 2);   //  8 MiB
  float*    bias  = (float*)   carve((size_t)R_DIM * 4);           // 16 KiB
  float*    Gx    = (float*)   carve((size_t)S_LEN * R_DIM * 4);   // 32 MiB
  uint16_t* hbuf  = (uint16_t*)carve((size_t)2 * H_DIM * 2);       // double-buffered h
  unsigned* cnt   = (unsigned*)carve(256);

  lstm_prep_weights<<<dim3((R_DIM * E_DIM) / 256), dim3(256), 0, stream>>>(
      Wih, Whh, b_ih, b_hh, Wihb, Whhb, bias);
  lstm_embed<<<dim3(S_LEN), dim3(256), 0, stream>>>(tokens, emb, Xbf);
  lstm_init<<<dim3(1), dim3(256), 0, stream>>>(hbuf, cnt);
  lstm_gemm_x<<<dim3(R_DIM / 256, S_LEN / 64), dim3(256), 0, stream>>>(
      Xbf, Wihb, bias, Gx);
  lstm_recurrent<<<dim3(NWG), dim3(256), 0, stream>>>(Whhb, Gx, hbuf, cnt, out);
}
